// AnatomaMamba_31301721653560
// MI455X (gfx1250) — compile-verified, matrix operand order checked by reading, not run on hardware
//
#include <hip/hip_runtime.h>
#include <hip/hip_bf16.h>
#include <math.h>

// ---------------------------------------------------------------------------
// MI455X (gfx1250) forward pass: Mamba + cross-attn multimodal LM.
// GEMM: bf16 tiles in LDS (converted once at staging), fragments via
// contiguous ds_load_b128, 8x v_wmma_f32_16x16x32_bf16 per wave per K-step.
// Async-to-LDS (ASYNCcnt) staging used in layernorm / attention row loads.
// ---------------------------------------------------------------------------

typedef __attribute__((ext_vector_type(16))) __bf16 v16bf;
typedef __attribute__((ext_vector_type(8)))  __bf16 v8bf;
typedef __attribute__((ext_vector_type(4)))  __bf16 v4bf;
typedef __attribute__((ext_vector_type(8)))  float  v8f;

#define DEV __device__ __forceinline__

DEV void async_ld_b128(unsigned ldsOff, const float* g) {
  asm volatile("global_load_async_to_lds_b128 %0, %1, off"
               :: "v"(ldsOff), "v"(g) : "memory");
}
DEV void wait_async0() {
  asm volatile("s_wait_asynccnt 0" ::: "memory");
}
DEV unsigned lds_off(const void* p) { return (unsigned)(size_t)p; }

DEV v4bf cvt4(float4 v) {
  v4bf r;
  r[0] = (__bf16)v.x; r[1] = (__bf16)v.y; r[2] = (__bf16)v.z; r[3] = (__bf16)v.w;
  return r;
}

// Build a 16-element bf16 fragment from a bf16 LDS row (ISA 16-bit 16x32
// layout): elements 0..7 = k[8*lg .. 8*lg+7], 8..15 = k[16+8*lg ..] --
// two contiguous 16B ds_load_b128 chunks.
DEV v16bf frag_ld(const __bf16* rowBase, int lg) {
  v8bf lo = *(const v8bf*)(rowBase + 8 * lg);
  v8bf hi = *(const v8bf*)(rowBase + 16 + 8 * lg);
  return __builtin_shufflevector(lo, hi, 0, 1, 2, 3, 4, 5, 6, 7,
                                 8, 9, 10, 11, 12, 13, 14, 15);
}

// ---------------------------------------------------------------------------
// WMMA GEMM: C[M,N] = (accum ? C : 0) + A[M,K] * op(B) + bias
//   op(B) = TRANSB ? B[N,K]^T : B[K,N]; f32 in memory, bf16 in the MACs.
// Block: 256 threads (8 waves) -> tile 64(M) x 128(N), K-step 64.
// Wave (wm 0..1, wn 0..3) computes 32x32 = 2x2 WMMA tiles x 2 k-halves.
// B always staged into LDS as [n][k] so A/B fragment gathers are identical.
// K is zero-padded at chunk granularity (handles K=32 with K-step 64).
// ---------------------------------------------------------------------------
#define GTM 64
#define GTN 128
#define GTK 64
#define AP16 72     // bf16 row pitch (64 k + pad), 144B = 16B-aligned
#define BP16 72

template <int TRANSB>
__global__ __launch_bounds__(256) void k_gemm(
    const float* __restrict__ A, int lda,
    const float* __restrict__ Bm, int ldb,
    const float* __restrict__ bias,
    float* __restrict__ C, int ldc,
    int M, int N, int K, int accum)
{
  __shared__ __align__(16) __bf16 As16[GTM * AP16];   //  9216 B
  __shared__ __align__(16) __bf16 Bt16[GTN * BP16];   // 18432 B

  const int tid  = threadIdx.x;
  const int lane = tid & 31;
  const int wid  = tid >> 5;
  const int wm   = wid & 1;        // M half
  const int wn   = wid >> 1;       // N quarter
  const int bm   = blockIdx.y * GTM;
  const int bn   = blockIdx.x * GTN;
  const int lg   = lane >> 4;
  const int l15  = lane & 15;
  const bool fullM = (bm + GTM <= M);
  const bool fullN = (bn + GTN <= N);

  v8f acc[2][2];
#pragma unroll
  for (int i = 0; i < 2; ++i)
#pragma unroll
    for (int j = 0; j < 2; ++j)
      acc[i][j] = v8f{0.f, 0.f, 0.f, 0.f, 0.f, 0.f, 0.f, 0.f};

  const v4bf z4 = {(__bf16)0.f, (__bf16)0.f, (__bf16)0.f, (__bf16)0.f};

  for (int k0 = 0; k0 < K; k0 += GTK) {
    const bool fullK = (k0 + GTK <= K);

    // ---- stage A: 64 rows x 64 k, f32 float4 load -> bf16 v4bf store ----
    for (int i = tid; i < GTM * (GTK / 4); i += 256) {
      int r = i >> 4, k4 = (i & 15) << 2;
      v4bf o;
      if ((fullM || bm + r < M) && (fullK || k0 + k4 < K)) {
        float4 v = *(const float4*)(A + (size_t)(bm + r) * lda + k0 + k4);
        o = cvt4(v);
      } else {
        o = z4;
      }
      *(v4bf*)(&As16[r * AP16 + k4]) = o;
    }
    // ---- stage B as [n][k]: 128 n-rows x 64 k ----
    for (int i = tid; i < GTN * (GTK / 4); i += 256) {
      int n = i >> 4, k4 = (i & 15) << 2;
      int gn = bn + n;
      v4bf o;
      if ((fullN || gn < N) && (fullK || k0 + k4 < K)) {
        if (TRANSB) {
          float4 v = *(const float4*)(Bm + (size_t)gn * ldb + k0 + k4);
          o = cvt4(v);
        } else {
          const float* bp = Bm + (size_t)(k0 + k4) * ldb + gn;
          float4 v = make_float4(bp[0], bp[(size_t)ldb],
                                 bp[2 * (size_t)ldb], bp[3 * (size_t)ldb]);
          o = cvt4(v);
        }
      } else {
        o = z4;
      }
      *(v4bf*)(&Bt16[n * BP16 + k4]) = o;
    }
    __syncthreads();

    // ---- 2 k-halves x 2x2 WMMA tiles, fragments straight from LDS ----
#pragma unroll
    for (int kh = 0; kh < 2; ++kh) {
      v16bf a0 = frag_ld(&As16[(wm * 32 + l15) * AP16 + 32 * kh], lg);
      v16bf a1 = frag_ld(&As16[(wm * 32 + 16 + l15) * AP16 + 32 * kh], lg);
      v16bf b0 = frag_ld(&Bt16[(wn * 32 + l15) * BP16 + 32 * kh], lg);
      v16bf b1 = frag_ld(&Bt16[(wn * 32 + 16 + l15) * BP16 + 32 * kh], lg);
      acc[0][0] = __builtin_amdgcn_wmma_f32_16x16x32_bf16(
          false, a0, false, b0, (short)0, acc[0][0], false, false);
      acc[0][1] = __builtin_amdgcn_wmma_f32_16x16x32_bf16(
          false, a0, false, b1, (short)0, acc[0][1], false, false);
      acc[1][0] = __builtin_amdgcn_wmma_f32_16x16x32_bf16(
          false, a1, false, b0, (short)0, acc[1][0], false, false);
      acc[1][1] = __builtin_amdgcn_wmma_f32_16x16x32_bf16(
          false, a1, false, b1, (short)0, acc[1][1], false, false);
    }
    __syncthreads();
  }

  // ---- epilogue: row = r + 8*lg, col = l15 within each 16x16 tile ----
#pragma unroll
  for (int i = 0; i < 2; ++i) {
#pragma unroll
    for (int j = 0; j < 2; ++j) {
      int nOut = bn + wn * 32 + j * 16 + l15;
#pragma unroll
      for (int r = 0; r < 8; ++r) {
        int mOut = bm + wm * 32 + i * 16 + r + 8 * lg;
        if (mOut < M && nOut < N) {
          float v = acc[i][j][r];
          if (bias) v += bias[nOut];
          size_t off = (size_t)mOut * ldc + nOut;
          if (accum) v += C[off];
          C[off] = v;
        }
      }
    }
  }
}

// ---------------------------------------------------------------------------
// Elementwise / normalization / scan / attention kernels
// ---------------------------------------------------------------------------

__global__ void k_dyt(const float* __restrict__ img, const float* __restrict__ alpha,
                      const float* __restrict__ g, const float* __restrict__ b,
                      float* __restrict__ out, int total, int Cd) {
  int idx = blockIdx.x * 256 + threadIdx.x;
  if (idx >= total) return;
  int c = idx % Cd;
  out[idx] = g[c] * tanhf(alpha[0] * img[idx]) + b[c];
}

__global__ void k_mean(const float* __restrict__ imgF, float* __restrict__ att0) {
  int idx = blockIdx.x * 256 + threadIdx.x;
  if (idx >= 8 * 512) return;
  int b = idx / 512, c = idx % 512;
  float s = 0.f;
  for (int j = 0; j < 196; ++j) s += imgF[(size_t)(b * 196 + j) * 512 + c];
  att0[idx] = s / 196.f;
}

__global__ void k_gelu(float* __restrict__ x, int n) {
  int i = blockIdx.x * 256 + threadIdx.x;
  if (i >= n) return;
  float v = x[i];
  x[i] = 0.5f * v * (1.f + erff(v * 0.70710678118654752f));
}

__global__ void k_sigmoid(float* __restrict__ x, int n) {
  int i = blockIdx.x * 256 + threadIdx.x;
  if (i >= n) return;
  x[i] = 1.f / (1.f + expf(-x[i]));
}

__global__ void k_softplus(float* __restrict__ x, int n) {
  int i = blockIdx.x * 256 + threadIdx.x;
  if (i >= n) return;
  float v = x[i];
  x[i] = (v > 20.f) ? v : log1pf(expf(v));
}

__global__ void k_scaleimg(const float* __restrict__ imgF, const float* __restrict__ att2,
                           float* __restrict__ kv, int total) {
  int idx = blockIdx.x * 256 + threadIdx.x;
  if (idx >= total) return;
  int c = idx % 512;
  int i = idx / 512;
  int b = i / 196;
  kv[idx] = imgF[idx] * att2[b * 512 + c];
}

__global__ void k_embed(const int* __restrict__ text, const float* __restrict__ tok,
                        const float* __restrict__ pos, float* __restrict__ x, int total) {
  int idx = blockIdx.x * 256 + threadIdx.x;
  if (idx >= total) return;
  int c = idx % 512;
  int t = idx / 512;
  int n = t % 256;
  int id = text[t];
  x[idx] = tok[(size_t)id * 512 + c] + pos[(size_t)n * 512 + c];
}

// LayerNorm, row staged into LDS with async-to-LDS b128 (ASYNCcnt path).
__global__ __launch_bounds__(256) void k_layernorm(
    const float* __restrict__ in, const float* __restrict__ g,
    const float* __restrict__ b, float* __restrict__ out, int D) {
  int row = blockIdx.x;
  const float* x = in + (size_t)row * D;
  __shared__ __align__(16) float xrow[512];
  __shared__ float red[256];
  for (int c = threadIdx.x * 4; c < D; c += 1024)
    async_ld_b128(lds_off(&xrow[c]), x + c);
  wait_async0();
  __syncthreads();

  float s = 0.f;
  for (int c = threadIdx.x; c < D; c += 256) s += xrow[c];
  red[threadIdx.x] = s; __syncthreads();
  for (int st = 128; st > 0; st >>= 1) {
    if (threadIdx.x < st) red[threadIdx.x] += red[threadIdx.x + st];
    __syncthreads();
  }
  float mean = red[0] / D; __syncthreads();
  float v = 0.f;
  for (int c = threadIdx.x; c < D; c += 256) { float d = xrow[c] - mean; v += d * d; }
  red[threadIdx.x] = v; __syncthreads();
  for (int st = 128; st > 0; st >>= 1) {
    if (threadIdx.x < st) red[threadIdx.x] += red[threadIdx.x + st];
    __syncthreads();
  }
  float inv = rsqrtf(red[0] / D + 1e-5f);
  float* o = out + (size_t)row * D;
  for (int c = threadIdx.x; c < D; c += 256)
    o[c] = (xrow[c] - mean) * inv * g[c] + b[c];
}

// Causal depthwise conv (D_CONV=4) + SiLU. xi = xz[:, :1024], row stride 2048.
__global__ void k_conv(const float* __restrict__ xz, const float* __restrict__ cW,
                       const float* __restrict__ cb, float* __restrict__ xc, int total) {
  int idx = blockIdx.x * 256 + threadIdx.x;
  if (idx >= total) return;
  int d = idx % 1024;
  int t = idx / 1024;
  int n = t % 256;
  float acc = cb[d];
#pragma unroll
  for (int k = 0; k < 4; ++k) {
    int nn = n + k - 3;
    if (nn >= 0) acc += xz[(size_t)(t + k - 3) * 2048 + d] * cW[d * 4 + k];
  }
  xc[idx] = acc / (1.f + expf(-acc));   // acc * sigmoid(acc)
}

// Selective scan: one thread per (b, d) channel; 16-state recurrence in regs.
__global__ __launch_bounds__(256) void k_scan(
    const float* __restrict__ dt, const float* __restrict__ xc,
    const float* __restrict__ dbl, const float* __restrict__ xz,
    const float* __restrict__ A_log, const float* __restrict__ Dsk,
    float* __restrict__ yw) {
  int idx = blockIdx.x * 256 + threadIdx.x;
  if (idx >= 8 * 1024) return;
  int d = idx % 1024;
  int b = idx / 1024;
  float Arow[16], h[16];
#pragma unroll
  for (int s = 0; s < 16; ++s) { Arow[s] = -expf(A_log[d * 16 + s]); h[s] = 0.f; }
  float dskip = Dsk[d];
  for (int n = 0; n < 256; ++n) {
    size_t t = (size_t)b * 256 + n;
    float dtv = dt[t * 1024 + d];
    float xcv = xc[t * 1024 + d];
    const float* Bp = dbl + t * 64 + 32;
    const float* Cp = dbl + t * 64 + 48;
    float y = 0.f;
#pragma unroll
    for (int s = 0; s < 16; ++s) {
      float dA = expf(dtv * Arow[s]);
      h[s] = h[s] * dA + dtv * Bp[s] * xcv;
      y += h[s] * Cp[s];
    }
    float zv = xz[t * 2048 + 1024 + d];
    float gate = zv / (1.f + expf(-zv));
    yw[t * 1024 + d] = (y + xcv * dskip) * gate;
  }
}

// Cross-attention: one block per (b, h, q). 196 keys, head_dim 64, scale 1/8.
__global__ __launch_bounds__(256) void k_attn(
    const float* __restrict__ q, const float* __restrict__ kb,
    const float* __restrict__ vb, float* __restrict__ o) {
  const int NK = 196, HD = 64, H = 8, NQ = 256;
  int idx = blockIdx.x;
  int qi = idx % NQ;
  int bh = idx / NQ;
  int h = bh % H;
  int b = bh / H;
  size_t t = (size_t)b * NQ + qi;
  __shared__ __align__(16) float qs[64];
  __shared__ float ps[224];
  __shared__ float red[256];
  if (threadIdx.x < 16)   // stage q row via async-to-LDS (16 x b128)
    async_ld_b128(lds_off(&qs[threadIdx.x * 4]),
                  q + t * 512 + h * HD + threadIdx.x * 4);
  wait_async0();
  __syncthreads();
  float sc = -1e30f;
  if ((int)threadIdx.x < NK) {
    const float* kp = kb + (size_t)(b * NK + threadIdx.x) * 512 + h * HD;
    float s = 0.f;
    for (int j = 0; j < HD; ++j) s += qs[j] * kp[j];
    sc = s * 0.125f;
  }
  red[threadIdx.x] = sc; __syncthreads();
  for (int st = 128; st > 0; st >>= 1) {
    if (threadIdx.x < st) red[threadIdx.x] = fmaxf(red[threadIdx.x], red[threadIdx.x + st]);
    __syncthreads();
  }
  float mx = red[0]; __syncthreads();
  float ex = 0.f;
  if ((int)threadIdx.x < NK) { ex = expf(sc - mx); ps[threadIdx.x] = ex; }
  red[threadIdx.x] = ex; __syncthreads();
  for (int st = 128; st > 0; st >>= 1) {
    if (threadIdx.x < st) red[threadIdx.x] += red[threadIdx.x + st];
    __syncthreads();
  }
  float denom = red[0]; __syncthreads();
  if (threadIdx.x < HD) {
    float acc = 0.f;
    for (int kk = 0; kk < NK; ++kk)
      acc += ps[kk] * vb[(size_t)(b * NK + kk) * 512 + h * HD + threadIdx.x];
    o[t * 512 + h * HD + threadIdx.x] = acc / denom;
  }
}

// ---------------------------------------------------------------------------
// Host orchestration
// ---------------------------------------------------------------------------
static inline int cdiv(int a, int b) { return (a + b - 1) / b; }

extern "C" void kernel_launch(void* const* d_in, const int* in_sizes, int n_in,
                              void* d_out, int out_size, void* d_ws, size_t ws_size,
                              hipStream_t stream) {
  (void)in_sizes; (void)n_in; (void)out_size; (void)ws_size;

  const int*   text      = (const int*)d_in[0];
  const float* images    = (const float*)d_in[1];
  const float* dyt_alpha = (const float*)d_in[2];
  const float* dyt_gamma = (const float*)d_in[3];
  const float* dyt_beta  = (const float*)d_in[4];
  const float* img_W     = (const float*)d_in[5];
  const float* img_b     = (const float*)d_in[6];
  const float* gate1_W   = (const float*)d_in[7];
  const float* gate1_b   = (const float*)d_in[8];
  const float* gate2_W   = (const float*)d_in[9];
  const float* gate2_b   = (const float*)d_in[10];
  const float* token_emb = (const float*)d_in[11];
  const float* pos_emb   = (const float*)d_in[12];
  const float* ln1_g     = (const float*)d_in[13];
  const float* ln1_b     = (const float*)d_in[14];
  const float* in_proj_W = (const float*)d_in[15];
  const float* conv_W    = (const float*)d_in[16];
  const float* conv_b    = (const float*)d_in[17];
  const float* x_proj_W  = (const float*)d_in[18];
  const float* dt_proj_W = (const float*)d_in[19];
  const float* dt_proj_b = (const float*)d_in[20];
  const float* A_log     = (const float*)d_in[21];
  const float* D_skip    = (const float*)d_in[22];
  const float* out_proj_W= (const float*)d_in[23];
  const float* ln2_g     = (const float*)d_in[24];
  const float* ln2_b     = (const float*)d_in[25];
  const float* attn_in_W = (const float*)d_in[26];
  const float* attn_in_b = (const float*)d_in[27];
  const float* attn_out_W= (const float*)d_in[28];
  const float* attn_out_b= (const float*)d_in[29];
  const float* fnorm_g   = (const float*)d_in[30];
  const float* fnorm_b   = (const float*)d_in[31];
  const float* logits_W  = (const float*)d_in[32];
  const float* logits_b  = (const float*)d_in[33];

  const int T = 2048;      // B*N text tokens
  const int I = 1568;      // B*CTX image tokens

  // workspace layout (floats), ~69 MB total
  float* w = (float*)d_ws;
  size_t off = 0;
  auto alloc = [&](size_t n) { float* p = w + off; off += n; return p; };
  float* kvbuf = alloc((size_t)I * 512);
  float* x     = alloc((size_t)T * 512);
  float* xn    = alloc((size_t)T * 512);
  float* xz    = alloc((size_t)T * 2048);
  float* xc    = alloc((size_t)T * 1024);
  float* dbl   = alloc((size_t)T * 64);
  float* dtb   = alloc((size_t)T * 1024);
  float* yw    = alloc((size_t)T * 1024);
  float* qb    = alloc((size_t)T * 512);
  float* kb    = alloc((size_t)I * 512);
  float* vb    = alloc((size_t)I * 512);
  float* ob    = alloc((size_t)T * 512);
  // phase-A aliases (regions unused until the layer loop)
  float* imgDyT = xz;            // I*1024 <= T*2048
  float* imgF   = xc;            // I*512  <= T*1024
  float* att0   = dbl;           // 8*512
  float* att1   = dbl + 8192;    // 8*128
  float* att2   = dbl + 16384;   // 8*512

  dim3 blk(256);
  auto gemm = [&](const float* A, int lda, const float* Bm, int ldb,
                  const float* bias, float* C, int ldc,
                  int M, int N, int K, int tB, int acc) {
    dim3 g(cdiv(N, GTN), cdiv(M, GTM));
    if (tB)
      k_gemm<1><<<g, blk, 0, stream>>>(A, lda, Bm, ldb, bias, C, ldc, M, N, K, acc);
    else
      k_gemm<0><<<g, blk, 0, stream>>>(A, lda, Bm, ldb, bias, C, ldc, M, N, K, acc);
  };

  // ---- image path ----
  int totImg = I * 1024;
  k_dyt<<<cdiv(totImg, 256), blk, 0, stream>>>(images, dyt_alpha, dyt_gamma, dyt_beta,
                                               imgDyT, totImg, 1024);
  gemm(imgDyT, 1024, img_W, 512, img_b, imgF, 512, I, 512, 1024, 0, 0);
  k_mean<<<cdiv(8 * 512, 256), blk, 0, stream>>>(imgF, att0);
  gemm(att0, 512, gate1_W, 128, gate1_b, att1, 128, 8, 128, 512, 0, 0);
  k_gelu<<<cdiv(8 * 128, 256), blk, 0, stream>>>(att1, 8 * 128);
  gemm(att1, 128, gate2_W, 512, gate2_b, att2, 512, 8, 512, 128, 0, 0);
  k_sigmoid<<<cdiv(8 * 512, 256), blk, 0, stream>>>(att2, 8 * 512);
  k_scaleimg<<<cdiv(I * 512, 256), blk, 0, stream>>>(imgF, att2, kvbuf, I * 512);

  // ---- embeddings ----
  k_embed<<<cdiv(T * 512, 256), blk, 0, stream>>>(text, token_emb, pos_emb, x, T * 512);

  // ---- layers ----
  for (int l = 0; l < 6; ++l) {
    k_layernorm<<<T, blk, 0, stream>>>(x, ln1_g + l * 512, ln1_b + l * 512, xn, 512);
    gemm(xn, 512, in_proj_W + (size_t)l * 512 * 2048, 2048, nullptr, xz, 2048,
         T, 2048, 512, 0, 0);
    k_conv<<<cdiv(T * 1024, 256), blk, 0, stream>>>(xz, conv_W + (size_t)l * 1024 * 4,
                                                    conv_b + l * 1024, xc, T * 1024);
    gemm(xc, 1024, x_proj_W + (size_t)l * 1024 * 64, 64, nullptr, dbl, 64,
         T, 64, 1024, 0, 0);
    gemm(dbl, 64, dt_proj_W + (size_t)l * 32 * 1024, 1024, dt_proj_b + l * 1024,
         dtb, 1024, T, 1024, 32, 0, 0);
    k_softplus<<<cdiv(T * 1024, 256), blk, 0, stream>>>(dtb, T * 1024);
    k_scan<<<cdiv(8 * 1024, 256), blk, 0, stream>>>(dtb, xc, dbl, xz,
                                                    A_log + (size_t)l * 1024 * 16,
                                                    D_skip + l * 1024, yw);
    gemm(yw, 1024, out_proj_W + (size_t)l * 1024 * 512, 512, nullptr, x, 512,
         T, 512, 1024, 0, 1);  // residual accumulate into x

    k_layernorm<<<T, blk, 0, stream>>>(x, ln2_g + l * 512, ln2_b + l * 512, xn, 512);
    const float* Wqkv = attn_in_W + (size_t)l * 1536 * 512;
    const float* bqkv = attn_in_b + (size_t)l * 1536;
    gemm(xn,    512, Wqkv,                 512, bqkv,        qb, 512, T, 512, 512, 1, 0);
    gemm(kvbuf, 512, Wqkv + 512 * 512,     512, bqkv + 512,  kb, 512, I, 512, 512, 1, 0);
    gemm(kvbuf, 512, Wqkv + 1024 * 512,    512, bqkv + 1024, vb, 512, I, 512, 512, 1, 0);
    k_attn<<<8 * 8 * 256, blk, 0, stream>>>(qb, kb, vb, ob);
    gemm(ob, 512, attn_out_W + (size_t)l * 512 * 512, 512, attn_out_b + l * 512,
         x, 512, T, 512, 512, 1, 1);  // residual accumulate into x
  }

  // ---- final norm + logits ----
  k_layernorm<<<T, blk, 0, stream>>>(x, fnorm_g, fnorm_b, xn, 512);
  gemm(xn, 512, logits_W, 10000, logits_b, (float*)d_out, 10000, T, 10000, 512, 0, 0);
}
